// NN_83330955478084
// MI455X (gfx1250) — matrix operand order repaired and checked by reading, not compiled, not run on hardware
//
#include <hip/hip_runtime.h>
#include <cstdint>
#include <cstddef>

// Problem constants (match reference)
#define B_   4096
#define T_   512
#define IN_  25
#define H_   50
#define G4_  (4 * H_)       // 200 real gate columns
#define NFRAG 48            // 16 wi tiles + 2*16 wh tiles (A fragments)
#define HPITCH 40           // h-exchange LDS pitch in dwords (16B aligned, bank-spread)

typedef __attribute__((ext_vector_type(16))) __bf16       v16bf;
typedef __attribute__((ext_vector_type(8)))  float        v8f;
typedef __attribute__((ext_vector_type(8)))  unsigned int v8u;

// ---------- scalar helpers ----------
__device__ __forceinline__ unsigned short f2bf(float f) {
  unsigned u = __float_as_uint(f);
  u += 0x7FFFu + ((u >> 16) & 1u);          // round-to-nearest-even
  return (unsigned short)(u >> 16);
}
__device__ __forceinline__ unsigned pk2(float lo, float hi) {
  return (unsigned)f2bf(lo) | ((unsigned)f2bf(hi) << 16);
}
__device__ __forceinline__ float bf2f(__bf16 h) {
  unsigned short s; __builtin_memcpy(&s, &h, 2);
  return __uint_as_float((unsigned)s << 16);
}
__device__ __forceinline__ float rcp_(float x) {       // v_rcp_f32
  return __builtin_amdgcn_rcpf(x);
}
__device__ __forceinline__ float sigm(float x) {       // exp + add + rcp
  return rcp_(1.0f + __expf(-x));
}
__device__ __forceinline__ float tanh_f(float x) {     // 2/(1+e^-2x)-1
  return fmaf(2.0f, rcp_(1.0f + __expf(-2.0f * x)), -1.0f);
}

// =====================================================================
// Kernel 1: build bf16 A-matrix fragments of W^T (gate-major padded) in ws.
// Fragment f, lane l, half h encodes element (M = gate-col-in-tile, K):
//   K order used here for A and B (contiguous): K = 16*(l>=16) + h
// Fragments: f in [0,16)   -> wi^T tile f   (K = input feature, bias in K==25)
//            f in [16,32)  -> wh^T tile, K = 0..31   (hidden cols 0..31)
//            f in [32,48)  -> wh^T tile, K = 32..63  (hidden cols 32..63)
// Padded gate column gc = tile*16 + M;  gate = gc>>6; idx = gc&63;
// real column = gate*50 + idx (valid iff idx < 50).
// =====================================================================
__global__ __launch_bounds__(256) void prep_weights(
    const float* __restrict__ wi_mu, const float* __restrict__ wi_rho,
    const float* __restrict__ wh_mu, const float* __restrict__ wh_rho,
    const float* __restrict__ b_mu,  const float* __restrict__ b_rho,
    const float* __restrict__ eps_wi, const float* __restrict__ eps_wh,
    const float* __restrict__ eps_b, unsigned short* __restrict__ W)
{
  int e = blockIdx.x * 256 + threadIdx.x;
  if (e >= NFRAG * 32 * 16) return;
  int f    = e >> 9;          // fragment (lane-major within fragment)
  int lane = (e >> 4) & 31;
  int h    = e & 15;
  int M    = lane & 15;
  int kloc = ((lane >= 16) ? 16 : 0) + h;

  int tile = (f < 16) ? f : ((f - 16) & 15);
  int gc   = tile * 16 + M;
  int gate = gc >> 6;
  int idx  = gc & 63;

  float val = 0.0f;
  if (idx < H_) {
    int col = gate * H_ + idx;                 // 0..199
    if (f < 16) {
      int K = kloc;                            // input feature (padded to 32)
      if (K < IN_) {
        int o = K * G4_ + col;
        val = wi_mu[o] + log1pf(expf(wi_rho[o])) * eps_wi[o];
      } else if (K == IN_) {                   // bias row (x slot 25 carries 1.0)
        val = b_mu[col] + log1pf(expf(b_rho[col])) * eps_b[col];
      }
    } else {
      int s = (f - 16) >> 4;                   // K-step 0/1
      int K = s * 32 + kloc;                   // hidden col (padded to 64)
      if (K < H_) {
        int o = K * G4_ + col;
        val = wh_mu[o] + log1pf(expf(wh_rho[o])) * eps_wh[o];
      }
    }
  }
  W[e] = f2bf(val);
}

// 3-WMMA gate accumulate from REGISTER-resident fragments (g = gate 0..3,
// must be a compile-time constant so wA[] indices stay constant).
__device__ __forceinline__ v8f mm3r(const v16bf* wA, int g,
                                    v16bf xb, v16bf hlo, v16bf hhi) {
  v8f acc = {};
  acc = __builtin_amdgcn_wmma_f32_16x16x32_bf16(
            false, wA[g],     false, xb,  (short)0, acc, false, false);
  acc = __builtin_amdgcn_wmma_f32_16x16x32_bf16(
            false, wA[4 + g], false, hlo, (short)0, acc, false, false);
  acc = __builtin_amdgcn_wmma_f32_16x16x32_bf16(
            false, wA[8 + g], false, hhi, (short)0, acc, false, false);
  return acc;
}

// Software-pipelined x row loader: raw fp32 features for one timestep.
// Lanes 0-15 consume fx[0..15] (features 0..15); lanes 16-31 consume
// fx[0..8] (features 16..24).
__device__ __forceinline__ void loadx(float* fx, const float* p, bool hiH) {
  if (!hiH) {
    __builtin_memcpy(fx, p, 16 * sizeof(float));
  } else {
    __builtin_memcpy(fx, p + 16, 9 * sizeof(float));
  }
}

// =====================================================================
// Kernel 2: fused LSTM scan + linear head.
// One workgroup (4 waves, one per SIMD of a WGP) = one 16-row batch tile.
// Wave j owns hidden-column tile j (cols 16j..16j+15): its i/f/g/o gate
// tiles (12 WMMAs/step) and 1/4 of the elementwise. The wave's 12 weight
// A-fragments are explicitly loaded into a register array ONCE before the
// scan (96 VGPRs; total <256 regs, no per-step LDS weight traffic). Each
// step the 4 waves exchange their 16 new h columns through a double-
// buffered LDS tile (aligned b128 DS ops) with ONE barrier per step.
// h-exchange layout: dword index = batch*HPITCH + colpair (colpair=col/2).
// =====================================================================
__global__ __launch_bounds__(128) void lstm_fused(
    const float* __restrict__ x, const unsigned short* __restrict__ Wsw,
    const float* __restrict__ lin_w, const float* __restrict__ lin_b,
    float* __restrict__ out)
{
  __shared__ v16bf    sW[NFRAG * 32];         // 48 KB of A-fragments (staging)
  __shared__ unsigned hx[2][16 * HPITCH];     // 5 KB double-buffered h exchange
  {
    const uint4* src = (const uint4*)Wsw;
    uint4* dst = (uint4*)sW;
    for (int i = threadIdx.x; i < NFRAG * 32 * 2; i += 128) dst[i] = src[i];
  }
  __syncthreads();

  const int  lane = threadIdx.x & 31;
  const int  j    = threadIdx.x >> 5;         // wave id = hidden-column tile
  const int  m0   = blockIdx.x * 16;          // batch tile base row
  const int  b    = lane & 15;
  const bool hiH  = lane >= 16;
  const float* xrow = x + (size_t)(m0 + b) * (T_ * IN_);

  // Pull this wave's 12 A-fragments into registers for the whole scan:
  // wA[g] = wi tile (gate g), wA[4+g] = wh K=0..31, wA[8+g] = wh K=32..63.
  v16bf wA[12];
#pragma unroll
  for (int g = 0; g < 4; ++g) {
    wA[g]     = sW[(g * 4 + j)      * 32 + lane];
    wA[4 + g] = sW[(16 + g * 4 + j) * 32 + lane];
    wA[8 + g] = sW[(32 + g * 4 + j) * 32 + lane];
  }

  // This wave's output columns: base..base+7 (pairs -> colpair base/2)
  const int cpStore = (j * 16 + (hiH ? 8 : 0)) >> 1;   // 0,4,...,28
  const int ldBase  = b * HPITCH;

  v16bf hlo = {}, hhi = {};                   // h^T B-fragments (cols 0..31 / 32..63)
  v8f   cj  = {};                             // this wave's c columns

  float fx[16];
  loadx(fx, xrow, hiH);                       // prime the pipeline (t = 0)

#pragma unroll 1
  for (int t = 0; t < T_; ++t) {
    // Build x^T B-fragment from the pre-loaded row: lanes 0-15 hold features
    // 0..15, lanes 16-31 hold features 16..31 (24 real, slot 25 = 1.0 bias).
    v8u xu = {};
    if (!hiH) {
#pragma unroll
      for (int k = 0; k < 8; ++k) xu[k] = pk2(fx[2 * k], fx[2 * k + 1]);
    } else {
#pragma unroll
      for (int k = 0; k < 4; ++k) xu[k] = pk2(fx[2 * k], fx[2 * k + 1]);
      xu[4] = pk2(fx[8], 1.0f);               // feature 24, bias-one at K=25
      xu[5] = 0u; xu[6] = 0u; xu[7] = 0u;
    }
    v16bf xb = __builtin_bit_cast(v16bf, xu);

    // Issue next step's loads now; latency hides under this step's compute.
    if (t + 1 < T_) {
      const float* pn = xrow + (t + 1) * IN_;
      if (t + 16 < T_) __builtin_prefetch(pn + 15 * IN_, 0, 0);
      loadx(fx, pn, hiH);
    }

    // Gates for this wave's 16 hidden columns: 12 WMMAs, operands all in regs.
    v8f ai = mm3r(wA, 0, xb, hlo, hhi);
    v8f af = mm3r(wA, 1, xb, hlo, hhi);
    v8f ag = mm3r(wA, 2, xb, hlo, hhi);
    v8f ao = mm3r(wA, 3, xb, hlo, hhi);

    float hn[8];
#pragma unroll
    for (int r = 0; r < 8; ++r) {
      float iv = sigm(ai[r]);
      float fv = sigm(af[r]);
      float gv = tanh_f(ag[r]);
      float ov = sigm(ao[r]);
      float cn = fmaf(fv, cj[r], iv * gv);
      cj[r] = cn;
      hn[r] = ov * tanh_f(cn);
    }

    // Exchange new h through LDS (double buffered, one barrier per step).
    const int buf = t & 1;
    uint4 pkv = make_uint4(pk2(hn[0], hn[1]), pk2(hn[2], hn[3]),
                           pk2(hn[4], hn[5]), pk2(hn[6], hn[7]));
    *(uint4*)&hx[buf][ldBase + cpStore] = pkv;        // 16B-aligned b128 store
    __syncthreads();

    const unsigned* srcp = &hx[buf][ldBase + (hiH ? 8 : 0)];
    uint4 l0 = *(const uint4*)(srcp);                 // cols (0|16) + 0..7
    uint4 l1 = *(const uint4*)(srcp + 4);
    uint4 h0 = *(const uint4*)(srcp + 16);            // cols (32|48) + 0..7
    uint4 h1 = *(const uint4*)(srcp + 20);
    v8u lo, hi;
    lo[0] = l0.x; lo[1] = l0.y; lo[2] = l0.z; lo[3] = l0.w;
    lo[4] = l1.x; lo[5] = l1.y; lo[6] = l1.z; lo[7] = l1.w;
    hi[0] = h0.x; hi[1] = h0.y; hi[2] = h0.z; hi[3] = h0.w;
    hi[4] = h1.x; hi[5] = h1.y; hi[6] = h1.z; hi[7] = h1.w;
    hlo = __builtin_bit_cast(v16bf, lo);
    hhi = __builtin_bit_cast(v16bf, hi);
  }

  // Linear head (wave 0 only): out[b] = sum_c h[b,c] * lin_w[c] + lin_b
  if (j == 0) {
    float sum = 0.0f;
    const int cb = hiH ? 16 : 0;
#pragma unroll
    for (int h = 0; h < 16; ++h) {
      int cl = cb + h;                        // <= 31 < 50, always valid
      int ch = 32 + cb + h;                   // may be padded col
      float wl = lin_w[cl];
      float wh = (ch < H_) ? lin_w[ch] : 0.0f;
      sum += bf2f(hlo[h]) * wl + bf2f(hhi[h]) * wh;
    }
    sum += __shfl_xor(sum, 16, 32);
    if (!hiH) out[m0 + b] = sum + lin_b[0];
  }
}

// =====================================================================
extern "C" void kernel_launch(void* const* d_in, const int* in_sizes, int n_in,
                              void* d_out, int out_size, void* d_ws, size_t ws_size,
                              hipStream_t stream) {
  (void)in_sizes; (void)n_in; (void)out_size; (void)ws_size;
  const float* x      = (const float*)d_in[0];
  const float* wi_mu  = (const float*)d_in[1];
  const float* wi_rho = (const float*)d_in[2];
  const float* wh_mu  = (const float*)d_in[3];
  const float* wh_rho = (const float*)d_in[4];
  const float* b_mu   = (const float*)d_in[5];
  const float* b_rho  = (const float*)d_in[6];
  const float* lin_w  = (const float*)d_in[7];
  const float* lin_b  = (const float*)d_in[8];
  const float* eps_wi = (const float*)d_in[9];
  const float* eps_wh = (const float*)d_in[10];
  const float* eps_b  = (const float*)d_in[11];

  unsigned short* W = (unsigned short*)d_ws;   // 48 KB of bf16 A-fragments

  const int nElem = NFRAG * 32 * 16;           // 24576
  prep_weights<<<(nElem + 255) / 256, 256, 0, stream>>>(
      wi_mu, wi_rho, wh_mu, wh_rho, b_mu, b_rho, eps_wi, eps_wh, eps_b, W);

  // 256 blocks x 4 waves (one per SIMD of a WGP): 1024 waves total,
  // one 16-row batch tile per block, gate columns split across waves.
  lstm_fused<<<256, 128, 0, stream>>>(x, W, lin_w, lin_b, (float*)d_out);
}
